// Qwen3OmniMoeMLP_23673859735580
// MI455X (gfx1250) — matrix operand order repaired and checked by reading, not compiled, not run on hardware
//
#include <hip/hip_runtime.h>

// ---------------- problem constants ----------------
#define Hd   1024
#define Id   2048
#define Ed   8
#define Nd   8192   // B*T = 4*2048

// ---------------- tile config ----------------
#define BM   128
#define BN   128
#define BK   32
#define LDSS 40       // padded row stride in bf16 elements (80B)
#define NK1  (Hd/BK)  // 32 k-steps for fc1
#define NK2  (Id/BK)  // 64 k-steps for fc2

typedef __attribute__((ext_vector_type(16))) __bf16 v16bf;
typedef __attribute__((ext_vector_type(8)))  __bf16 v8bf;
typedef __attribute__((ext_vector_type(4)))  __bf16 v4bf;
typedef __attribute__((ext_vector_type(8)))  float  v8f;

union BFrag { v16bf v; v8bf h[2]; };

// ---------------- CDNA5 async-to-LDS helpers ----------------
__device__ __forceinline__ void async_copy_b64(unsigned ldsoff, const void* g)
{
    // GLOBAL_LOAD_ASYNC_TO_LDS_B64: copies 8 bytes from global g to LDS offset ldsoff, ASYNCcnt-tracked
    asm volatile("global_load_async_to_lds_b64 %0, %1, off"
                 :: "v"(ldsoff), "v"(g) : "memory");
}
__device__ __forceinline__ void wait_async0()
{
    asm volatile("s_wait_asynccnt 0" ::: "memory");
}
__device__ __forceinline__ unsigned lds_addr32(const void* p)
{
    // flat shared pointer: aperture in [63:32], LDS byte offset in [31:0]
    return (unsigned)(size_t)p;
}

__device__ __forceinline__ void cvt_store(__bf16* dst, float4 v)
{
    v4bf t = { (__bf16)v.x, (__bf16)v.y, (__bf16)v.z, (__bf16)v.w };
    *(v4bf*)dst = t;
}
__device__ __forceinline__ float fast_silu(float v)
{
    return v * __builtin_amdgcn_rcpf(1.f + __expf(-v));
}

// =====================================================================
// Kernel 1: gating.  One wave (32 lanes) per token.
// =====================================================================
__global__ __launch_bounds__(256) void moe_gate_kernel(
    const float* __restrict__ x, const float* __restrict__ Wg,
    float* __restrict__ w)
{
    const int wave = threadIdx.x >> 5;
    const int lane = threadIdx.x & 31;
    const int n = blockIdx.x * 8 + wave;
    if (n >= Nd) return;

    const float* xr = x + (size_t)n * Hd;
    float acc[Ed];
#pragma unroll
    for (int e = 0; e < Ed; ++e) acc[e] = 0.f;

    for (int h = lane; h < Hd; h += 32) {
        const float xv = xr[h];
#pragma unroll
        for (int e = 0; e < Ed; ++e) acc[e] += xv * Wg[e * Hd + h];
    }
#pragma unroll
    for (int e = 0; e < Ed; ++e) {
#pragma unroll
        for (int off = 16; off > 0; off >>= 1)
            acc[e] += __shfl_xor(acc[e], off, 32);
    }
    if (lane == 0) {
        float m = acc[0];
#pragma unroll
        for (int e = 1; e < Ed; ++e) m = fmaxf(m, acc[e]);
        float p[Ed]; float s = 0.f;
#pragma unroll
        for (int e = 0; e < Ed; ++e) { p[e] = __expf(acc[e] - m); s += p[e]; }
        const float inv = 1.f / s;
        int i0 = 0;
#pragma unroll
        for (int e = 1; e < Ed; ++e) if (p[e] > p[i0]) i0 = e;
        int i1 = (i0 == 0) ? 1 : 0;
#pragma unroll
        for (int e = 0; e < Ed; ++e)
            if (e != i0 && e != i1 && p[e] > p[i1]) i1 = e;
#pragma unroll
        for (int e = 0; e < Ed; ++e)
            w[(size_t)n * Ed + e] = (e == i0 || e == i1) ? p[e] * inv : 0.f;
    }
}

// ---------------------------------------------------------------------
// LDS fragment gathers per CDNA5 ISA 7.12.2 (wave32, 16-bit operands).
// ---------------------------------------------------------------------
__device__ __forceinline__ v16bf load_a_frag(const __bf16* tile, int rowBase,
                                             int lrow, int lhalf)
{
    BFrag f;
    const __bf16* p = tile + (rowBase + lrow) * LDSS + lhalf * 8;
    f.h[0] = *(const v8bf*)(p);
    f.h[1] = *(const v8bf*)(p + 16);
    return f.v;
}
__device__ __forceinline__ v16bf load_b_frag(const __bf16* tile, int colBase,
                                             int lrow, int lhalf)
{
    BFrag f;
    const __bf16* p = tile + (colBase + lrow) * LDSS + lhalf * 8;
    f.h[0] = *(const v8bf*)(p);
    f.h[1] = *(const v8bf*)(p + 16);
    return f.v;
}

// =====================================================================
// Kernel 2: fc1 (one expert): hid[N,I] = silu(x @ W1e^T), bf16 out.
// Double-buffered LDS, register-prefetched globals, f32->bf16 on the fly.
// =====================================================================
__global__ __launch_bounds__(256) void moe_fc1_kernel(
    const float* __restrict__ x, const float* __restrict__ W1e,
    __bf16* __restrict__ hid)
{
    __shared__ __bf16 Als[2][BM * LDSS];
    __shared__ __bf16 Bls[2][BN * LDSS];

    const int tid  = threadIdx.x;
    const int m0   = blockIdx.y * BM;
    const int n0   = blockIdx.x * BN;
    const int wid  = tid >> 5;
    const int lane = tid & 31;
    const int wm   = (wid & 3) * 32;
    const int wn   = (wid >> 2) * 64;
    const int lrow = lane & 15;
    const int lhalf = lane >> 4;

    int srow[4], skc[4];
#pragma unroll
    for (int it = 0; it < 4; ++it) {
        const int g = tid + 256 * it;
        srow[it] = g >> 3; skc[it] = (g & 7) * 4;
    }
    const float* ap[4]; const float* bp[4];
#pragma unroll
    for (int it = 0; it < 4; ++it) {
        ap[it] = x   + (size_t)(m0 + srow[it]) * Hd + skc[it];
        bp[it] = W1e + (size_t)(n0 + srow[it]) * Hd + skc[it];
    }

    float4 ar[4], br[4];
    // prologue: stage k-step 0 into buffer 0
#pragma unroll
    for (int it = 0; it < 4; ++it) { ar[it] = *(const float4*)(ap[it]); br[it] = *(const float4*)(bp[it]); }
#pragma unroll
    for (int it = 0; it < 4; ++it) {
        cvt_store(&Als[0][srow[it] * LDSS + skc[it]], ar[it]);
        cvt_store(&Bls[0][srow[it] * LDSS + skc[it]], br[it]);
    }
    __syncthreads();

    v8f acc[2][4] = {};
    for (int t = 0; t < NK1; ++t) {
        const int cur = t & 1, nxt = cur ^ 1;
        if (t + 1 < NK1) {
            const int k0 = (t + 1) * BK;
#pragma unroll
            for (int it = 0; it < 4; ++it) {
                ar[it] = *(const float4*)(ap[it] + k0);
                br[it] = *(const float4*)(bp[it] + k0);
            }
            if (t + 3 < NK1) {
                __builtin_prefetch(ap[0] + (t + 3) * BK, 0, 1);
                __builtin_prefetch(bp[0] + (t + 3) * BK, 0, 1);
            }
        }
        v16bf a[2], b[4];
#pragma unroll
        for (int mt = 0; mt < 2; ++mt) a[mt] = load_a_frag(Als[cur], wm + mt * 16, lrow, lhalf);
#pragma unroll
        for (int nt = 0; nt < 4; ++nt) b[nt] = load_b_frag(Bls[cur], wn + nt * 16, lrow, lhalf);
#pragma unroll
        for (int mt = 0; mt < 2; ++mt)
#pragma unroll
            for (int nt = 0; nt < 4; ++nt)
                acc[mt][nt] = __builtin_amdgcn_wmma_f32_16x16x32_bf16(
                    false, a[mt], false, b[nt], (short)0, acc[mt][nt], false, false);
        if (t + 1 < NK1) {
#pragma unroll
            for (int it = 0; it < 4; ++it) {
                cvt_store(&Als[nxt][srow[it] * LDSS + skc[it]], ar[it]);
                cvt_store(&Bls[nxt][srow[it] * LDSS + skc[it]], br[it]);
            }
        }
        __syncthreads();
    }

#pragma unroll
    for (int mt = 0; mt < 2; ++mt) {
#pragma unroll
        for (int nt = 0; nt < 4; ++nt) {
            const int mbase = m0 + wm + mt * 16 + lhalf * 8;
            const int nn    = n0 + wn + nt * 16 + lrow;
#pragma unroll
            for (int r = 0; r < 8; ++r) {
                const float v = acc[mt][nt][r];
                hid[(size_t)(mbase + r) * Id + nn] = (__bf16)fast_silu(v);
            }
        }
    }
}

// =====================================================================
// Kernel 3: fc2 (one expert): out[n,h] (+)= w[n,e] * (hid @ W2e^T).
// A tile (bf16) staged with GLOBAL_LOAD_ASYNC_TO_LDS_B64; B converted f32->bf16.
// =====================================================================
__global__ __launch_bounds__(256) void moe_fc2_kernel(
    const __bf16* __restrict__ hid, const float* __restrict__ W2e,
    const float* __restrict__ wexp /* w + e, stride E */,
    float* __restrict__ out, int addTo)
{
    __shared__ __bf16 Als[2][BM * LDSS];
    __shared__ __bf16 Bls[2][BN * LDSS];

    const int tid  = threadIdx.x;
    const int m0   = blockIdx.y * BM;
    const int n0   = blockIdx.x * BN;
    const int wid  = tid >> 5;
    const int lane = tid & 31;
    const int wm   = (wid & 3) * 32;
    const int wn   = (wid >> 2) * 64;
    const int lrow = lane & 15;
    const int lhalf = lane >> 4;

    int srow[4], skc[4];
#pragma unroll
    for (int it = 0; it < 4; ++it) {
        const int g = tid + 256 * it;
        srow[it] = g >> 3; skc[it] = (g & 7) * 4;
    }
    unsigned aoff[2][4];
    const __bf16* asrc[4];
    const float*  bp[4];
#pragma unroll
    for (int it = 0; it < 4; ++it) {
        aoff[0][it] = lds_addr32(&Als[0][srow[it] * LDSS + skc[it]]);
        aoff[1][it] = lds_addr32(&Als[1][srow[it] * LDSS + skc[it]]);
        asrc[it] = hid + (size_t)(m0 + srow[it]) * Id + skc[it];
        bp[it]   = W2e + (size_t)(n0 + srow[it]) * Id + skc[it];
    }

    float4 br[4];
    // prologue: k-step 0 -> buffer 0 (A async, B via regs+cvt)
#pragma unroll
    for (int it = 0; it < 4; ++it) async_copy_b64(aoff[0][it], asrc[it]);
#pragma unroll
    for (int it = 0; it < 4; ++it) br[it] = *(const float4*)(bp[it]);
#pragma unroll
    for (int it = 0; it < 4; ++it) cvt_store(&Bls[0][srow[it] * LDSS + skc[it]], br[it]);
    wait_async0();
    __syncthreads();

    v8f acc[2][4] = {};
    for (int t = 0; t < NK2; ++t) {
        const int cur = t & 1, nxt = cur ^ 1;
        if (t + 1 < NK2) {
            const int k0 = (t + 1) * BK;
#pragma unroll
            for (int it = 0; it < 4; ++it) async_copy_b64(aoff[nxt][it], asrc[it] + k0);
#pragma unroll
            for (int it = 0; it < 4; ++it) br[it] = *(const float4*)(bp[it] + k0);
            if (t + 3 < NK2)
                __builtin_prefetch(bp[0] + (t + 3) * BK, 0, 1);
        }
        v16bf a[2], b[4];
#pragma unroll
        for (int mt = 0; mt < 2; ++mt) a[mt] = load_a_frag(Als[cur], wm + mt * 16, lrow, lhalf);
#pragma unroll
        for (int nt = 0; nt < 4; ++nt) b[nt] = load_b_frag(Bls[cur], wn + nt * 16, lrow, lhalf);
#pragma unroll
        for (int mt = 0; mt < 2; ++mt)
#pragma unroll
            for (int nt = 0; nt < 4; ++nt)
                acc[mt][nt] = __builtin_amdgcn_wmma_f32_16x16x32_bf16(
                    false, a[mt], false, b[nt], (short)0, acc[mt][nt], false, false);
        if (t + 1 < NK2) {
#pragma unroll
            for (int it = 0; it < 4; ++it)
                cvt_store(&Bls[nxt][srow[it] * LDSS + skc[it]], br[it]);
        }
        wait_async0();   // own async writes to buf[nxt] done before publishing barrier
        __syncthreads();
    }

#pragma unroll
    for (int mt = 0; mt < 2; ++mt) {
#pragma unroll
        for (int nt = 0; nt < 4; ++nt) {
            const int mbase = m0 + wm + mt * 16 + lhalf * 8;
            const int nn    = n0 + wn + nt * 16 + lrow;
#pragma unroll
            for (int r = 0; r < 8; ++r) {
                const int token = mbase + r;
                const float gw  = wexp[(size_t)token * Ed];
                const size_t oi = (size_t)token * Hd + nn;
                const float prev = addTo ? out[oi] : 0.f;
                out[oi] = prev + gw * acc[mt][nt][r];
            }
        }
    }
}

// =====================================================================
extern "C" void kernel_launch(void* const* d_in, const int* in_sizes, int n_in,
                              void* d_out, int out_size, void* d_ws, size_t ws_size,
                              hipStream_t stream)
{
    const float* x  = (const float*)d_in[0];   // [N,H]
    const float* Wg = (const float*)d_in[1];   // [E,H]
    const float* W1 = (const float*)d_in[2];   // [E,I,H]
    const float* W2 = (const float*)d_in[3];   // [E,H,I]
    float* out = (float*)d_out;                // [N,H]

    float*  w   = (float*)d_ws;                                  // [N,E]
    __bf16* hid = (__bf16*)((char*)d_ws + (size_t)Nd * Ed * 4);  // [N,I] bf16, reused per expert

    moe_gate_kernel<<<Nd / 8, 256, 0, stream>>>(x, Wg, w);

    const dim3 g1(Id / BN, Nd / BM);   // (16, 64)
    const dim3 g2(Hd / BN, Nd / BM);   // (8, 64)
    for (int e = 0; e < Ed; ++e) {
        moe_fc1_kernel<<<g1, 256, 0, stream>>>(
            x, W1 + (size_t)e * Id * Hd, hid);
        moe_fc2_kernel<<<g2, 256, 0, stream>>>(
            hid, W2 + (size_t)e * Hd * Id, w + e, out, e > 0);
    }
}
